// TransitionDown_25056839205738
// MI455X (gfx1250) — matrix operand order repaired and checked by reading, not compiled, not run on hardware
//
#include <hip/hip_runtime.h>

#define C_IN   64
#define C_OUT  128
#define KNN_K  16
#define LN_EPS 1e-5f

typedef __attribute__((ext_vector_type(16))) _Float16 v16h;
typedef __attribute__((ext_vector_type(8)))  float    v8f;
typedef unsigned int v4u __attribute__((ext_vector_type(4)));
typedef int          v4i __attribute__((ext_vector_type(4)));
typedef int          v8i __attribute__((ext_vector_type(8)));

// Swizzled W: 16 fragments (nt 0..7 x kc 0..1), each 32 lanes x 16 halves,
// contiguous per lane. Total 16*512 halves = 16 KB.
#define WSW_HALVES (16 * 512)

// ---------------------------------------------------------------------------
// Phase 1: furthest point sampling. Single 1024-thread workgroup; running
// min-distance array in ws (L2-resident); wave32 shuffle argmax + LDS tree.
// ---------------------------------------------------------------------------
__global__ __launch_bounds__(1024) void fps_kernel(
    const float* __restrict__ xyz, float* __restrict__ dists,
    float* __restrict__ n_xyz, int N, int M)
{
    __shared__ float red_d[32];
    __shared__ int   red_i[32];
    __shared__ int   s_next;
    __shared__ float s_px, s_py, s_pz;

    const int tid  = threadIdx.x;
    const int lane = tid & 31;
    const int wid  = tid >> 5;

    for (int i = tid; i < N; i += 1024) dists[i] = 1e10f;
    int last = 0;
    __syncthreads();

    for (int m = 0; m < M; ++m) {
        if (tid == 0) {
            float px = xyz[3*last], py = xyz[3*last+1], pz = xyz[3*last+2];
            s_px = px; s_py = py; s_pz = pz;
            n_xyz[3*m] = px; n_xyz[3*m+1] = py; n_xyz[3*m+2] = pz;
        }
        __syncthreads();
        const float px = s_px, py = s_py, pz = s_pz;

        float bmax = -1.0f; int bidx = 0x7fffffff;
        for (int i = tid; i < N; i += 1024) {
            float dx = xyz[3*i] - px, dy = xyz[3*i+1] - py, dz = xyz[3*i+2] - pz;
            float d  = dx*dx + dy*dy + dz*dz;
            float nd = fminf(dists[i], d);
            dists[i] = nd;
            if (nd > bmax) { bmax = nd; bidx = i; }
        }
        #pragma unroll
        for (int off = 16; off > 0; off >>= 1) {
            float od = __shfl_xor(bmax, off);
            int   oi = __shfl_xor(bidx, off);
            if (od > bmax || (od == bmax && oi < bidx)) { bmax = od; bidx = oi; }
        }
        if (lane == 0) { red_d[wid] = bmax; red_i[wid] = bidx; }
        __syncthreads();
        if (wid == 0) {
            float rb = red_d[lane]; int ri = red_i[lane];
            #pragma unroll
            for (int off = 16; off > 0; off >>= 1) {
                float od = __shfl_xor(rb, off);
                int   oi = __shfl_xor(ri, off);
                if (od > rb || (od == rb && oi < ri)) { rb = od; ri = oi; }
            }
            if (lane == 0) s_next = ri;
        }
        __syncthreads();
        last = s_next;
    }
}

// ---------------------------------------------------------------------------
// Phase 2: KNN. One wave per center; register-resident sorted top-16 per lane
// (static indexing only), then a 16-round wave argmin merge.
// ---------------------------------------------------------------------------
__global__ __launch_bounds__(256) void knn_kernel(
    const float* __restrict__ xyz, const float* __restrict__ n_xyz,
    int* __restrict__ knn, int N, int M)
{
    const int lane = threadIdx.x & 31;
    const int w    = (int)((blockIdx.x * blockDim.x + threadIdx.x) >> 5);
    if (w >= M) return;

    const float cx = n_xyz[3*w], cy = n_xyz[3*w+1], cz = n_xyz[3*w+2];

    float bd[KNN_K]; int bi[KNN_K];
    #pragma unroll
    for (int j = 0; j < KNN_K; ++j) { bd[j] = 3.0e38f; bi[j] = -1; }

    for (int i = lane; i < N; i += 32) {
        __builtin_prefetch(&xyz[3*(i + 256)], 0, 1);   // global_prefetch_b8
        float dx = xyz[3*i]-cx, dy = xyz[3*i+1]-cy, dz = xyz[3*i+2]-cz;
        float d = dx*dx + dy*dy + dz*dz;
        if (d < bd[KNN_K-1]) {
            #pragma unroll
            for (int j = KNN_K-1; j > 0; --j) {
                if (bd[j] > d) {
                    bool sh = bd[j-1] > d;
                    bd[j] = sh ? bd[j-1] : d;
                    bi[j] = sh ? bi[j-1] : i;
                }
            }
            if (bd[0] > d) { bd[0] = d; bi[0] = i; }
        }
    }

    #pragma unroll 1
    for (int r = 0; r < KNN_K; ++r) {
        float md = bd[0]; int mi = bi[0];
        #pragma unroll
        for (int off = 16; off > 0; off >>= 1) {
            float od = __shfl_xor(md, off);
            int   oi = __shfl_xor(mi, off);
            if (od < md || (od == md && oi < mi)) { md = od; mi = oi; }
        }
        if (lane == 0) knn[w*KNN_K + r] = mi;
        if (bi[0] == mi) {
            #pragma unroll
            for (int j = 0; j < KNN_K-1; ++j) { bd[j] = bd[j+1]; bi[j] = bi[j+1]; }
            bd[KNN_K-1] = 3.0e38f; bi[KNN_K-1] = -1;
        }
    }
}

// ---------------------------------------------------------------------------
// Pre-swizzle W (f32, 64x128) into f16 WMMA B-fragment order:
//   frag = nt*2 + kc ; within frag: lane l, element e -> K = (e>>3)*16 + (l>>4)*8 + (e&7),
//   row = kc*32 + K, col = nt*16 + (l&15).
// Thread 0 also writes the n_offset output.
// ---------------------------------------------------------------------------
__global__ void wswz_kernel(const float* __restrict__ W, _Float16* __restrict__ Wsw,
                            float* __restrict__ tail, int M)
{
    int i = blockIdx.x * blockDim.x + threadIdx.x;
    if (i < WSW_HALVES) {
        int frag = i >> 9;
        int rem  = i & 511;
        int l    = rem >> 4;
        int e    = rem & 15;
        int nt   = frag >> 1;
        int kc   = frag & 1;
        int K    = (e >> 3) * 16 + (l >> 4) * 8 + (e & 7);
        int row  = kc * 32 + K;
        int col  = nt * 16 + (l & 15);
        Wsw[i] = (_Float16)W[row * C_OUT + col];
    }
    if (i == 0) ((int*)tail)[0] = M;
}

// ---------------------------------------------------------------------------
// Phase 3: gather + LayerNorm + (16x64)x(64x128) WMMA + max over K neighbors.
// One wave per center. Swizzled W (16 KB) is DMA'd into LDS once per
// workgroup by the Tensor Data Mover; B fragments are then 32-byte LDS reads.
// ---------------------------------------------------------------------------
__global__ __launch_bounds__(256) void tdown_kernel(
    const float* __restrict__ feats, const int* __restrict__ knn,
    const float* __restrict__ nw, const float* __restrict__ nb,
    const _Float16* __restrict__ Wsw, float* __restrict__ out, int M)
{
    __shared__ __align__(32) _Float16 sW[WSW_HALVES];   // 16 KB

    // --- TDM: stage swizzled W into LDS (wave 0 issues; EXEC ignored) ------
    if (threadIdx.x < 32) {
        unsigned long long ga = (unsigned long long)(size_t)Wsw;
        unsigned ldsoff = (unsigned)(size_t)(&sW[0]);   // low 32 bits = LDS offset
        v4u g0 = { 1u,                                   // count=1 (valid D#)
                   ldsoff,                               // lds_addr
                   (unsigned)(ga & 0xffffffffu),         // global_addr[31:0]
                   (unsigned)((ga >> 32) & 0x01ffffffu)  // global_addr[56:32]
                     | 0x80000000u };                    // type=2 ("image")
        v8i g1 = { (int)0x00020000,            // data_size=2 (4B dwords)
                   (int)(4096u << 16),         // tensor_dim0 = 4096 (lo16)
                   (int)0x00010000,            // td0 hi16=0 | tensor_dim1=1
                   (int)(4096u << 16),         // td1 hi16=0 | tile_dim0=4096
                   1,                          // tile_dim1=1 | tile_dim2=0
                   4096,                       // tensor_dim0_stride lo32
                   0, 0 };                     // stride hi | dim1_stride
        v4i gz4 = { 0, 0, 0, 0 };              // groups 2/3 unused (<=2D)
        v8i gz8 = { 0, 0, 0, 0, 0, 0, 0, 0 };  // clang-23 6-arg form extra group
        __builtin_amdgcn_tensor_load_to_lds(g0, g1, gz4, gz4, gz8, 0);
        __builtin_amdgcn_s_wait_tensorcnt(0);
    }
    __syncthreads();

    const int lane = threadIdx.x & 31;
    const int w    = (int)((blockIdx.x * blockDim.x + threadIdx.x) >> 5);
    if (w >= M) return;                 // wave-uniform: EXEC all-ones below

    const int row  = lane & 15;         // A-matrix row (neighbor index)
    const int half = lane >> 4;         // K-half select
    const int nbr  = knn[w*KNN_K + row];
    const float*  f  = feats + (long)nbr * C_IN;
    const float4* f4 = (const float4*)f;       // rows are 256B-aligned

    // Gather this lane's 32 channels as 8x b128; accumulate LN statistics.
    float vals[32];
    float sum = 0.f, sq = 0.f;
    #pragma unroll
    for (int g = 0; g < 4; ++g) {
        float4 lo = f4[g*4 + half*2];
        float4 hi = f4[g*4 + half*2 + 1];
        vals[g*8+0] = lo.x; vals[g*8+1] = lo.y; vals[g*8+2] = lo.z; vals[g*8+3] = lo.w;
        vals[g*8+4] = hi.x; vals[g*8+5] = hi.y; vals[g*8+6] = hi.z; vals[g*8+7] = hi.w;
        sum += lo.x + lo.y + lo.z + lo.w + hi.x + hi.y + hi.z + hi.w;
        sq  += lo.x*lo.x + lo.y*lo.y + lo.z*lo.z + lo.w*lo.w
             + hi.x*hi.x + hi.y*hi.y + hi.z*hi.z + hi.w*hi.w;
    }
    sum += __shfl_xor(sum, 16);         // partner lane holds the other 32 ch
    sq  += __shfl_xor(sq, 16);
    const float mean = sum * (1.0f/64.0f);
    const float var  = sq  * (1.0f/64.0f) - mean*mean;
    const float rstd = rsqrtf(var + LN_EPS);

    // Pack A fragments (ISA 16-bit A 16x32 layout; identity on vals order).
    v16h a0, a1;
    #pragma unroll
    for (int e = 0; e < 16; ++e) {
        int g0i = e >> 3, o0 = e & 7;
        int ch0 = g0i*16 + half*8 + o0;          // K-chunk 0 channel
        int ch1 = (g0i+2)*16 + half*8 + o0;      // K-chunk 1 channel
        a0[e] = (_Float16)((vals[e]    - mean) * rstd * nw[ch0] + nb[ch0]);
        a1[e] = (_Float16)((vals[16+e] - mean) * rstd * nw[ch1] + nb[ch1]);
    }

    float* outp = out + (long)w * C_OUT;

    #pragma unroll 1
    for (int nt = 0; nt < 8; ++nt) {
        const int fb = (nt << 1) * 512 + lane * 16;   // frag (nt,kc=0)
        v16h b0 = *(const v16h*)&sW[fb];
        v16h b1 = *(const v16h*)&sW[fb + 512];        // frag (nt,kc=1)
        v8f acc = {};
        acc = __builtin_amdgcn_wmma_f32_16x16x32_f16(false, a0, false, b0,
                                                     (short)0, acc, false, false);
        acc = __builtin_amdgcn_wmma_f32_16x16x32_f16(false, a1, false, b1,
                                                     (short)0, acc, false, false);
        // Max-pool over 16 neighbor rows: 8 acc VGPRs = rows (half*8 + r).
        float m = acc[0];
        #pragma unroll
        for (int r = 1; r < 8; ++r) m = fmaxf(m, acc[r]);
        m = fmaxf(m, __shfl_xor(m, 16));
        if (lane < 16) outp[nt*16 + lane] = m;
    }
}

// ---------------------------------------------------------------------------
extern "C" void kernel_launch(void* const* d_in, const int* in_sizes, int n_in,
                              void* d_out, int out_size, void* d_ws, size_t ws_size,
                              hipStream_t stream) {
    const float* feats = (const float*)d_in[0];
    const float* xyz   = (const float*)d_in[1];
    // d_in[2] = offset (single point cloud; unused)
    const float* nw    = (const float*)d_in[3];
    const float* nb    = (const float*)d_in[4];
    const float* W     = (const float*)d_in[5];

    const int N = in_sizes[1] / 3;
    const int M = N / 4 + 1;            // int(N*0.25)+1

    float* out   = (float*)d_out;                    // [M, 128]
    float* n_xyz = out + (long)M * C_OUT;            // [M, 3]
    float* tail  = n_xyz + (long)M * 3;              // n_offset (int bits)

    char* ws = (char*)d_ws;
    size_t off1 = (((size_t)N * 4 + 255) & ~(size_t)255);                // dists end
    size_t off2 = off1 + (((size_t)M * KNN_K * 4 + 255) & ~(size_t)255); // knn end
    float*    dists = (float*)ws;
    int*      knn   = (int*)(ws + off1);
    _Float16* Wsw   = (_Float16*)(ws + off2);        // 16 KB, 256B-aligned

    fps_kernel<<<1, 1024, 0, stream>>>(xyz, dists, n_xyz, N, M);
    wswz_kernel<<<(WSW_HALVES + 255) / 256, 256, 0, stream>>>(W, Wsw, tail, M);

    const int blocks = (M * 32 + 255) / 256;         // one wave per center
    knn_kernel<<<blocks, 256, 0, stream>>>(xyz, n_xyz, knn, N, M);
    tdown_kernel<<<blocks, 256, 0, stream>>>(feats, knn, nw, nb, Wsw, out, M);
}
